// ConditionalAttention_12103217840438
// MI455X (gfx1250) — matrix operand order repaired and checked by reading, not compiled, not run on hardware
//
#include <hip/hip_runtime.h>

// ---------------------------------------------------------------------------
// CDNA5 (gfx1250) routed-attention pipeline, wave32 + WMMA bf16.
// Round 3: fix async builtin arg types — expects (as1 int4-vector*, as3
// int4-vector*, imm offset, imm cpol) per clang diagnostic.
// ---------------------------------------------------------------------------

typedef __attribute__((ext_vector_type(16))) __bf16 v16bf;
typedef __attribute__((ext_vector_type(8)))  float  v8f;
typedef unsigned int   u32;
typedef unsigned short bf16_t;
typedef int gvec4 __attribute__((vector_size(16)));   // V4i for async builtins

#define DIM_B   4
#define DIM_N   8192
#define DIM_D   1024
#define DIM_H   16
#define DIM_DH  64
#define DIM_NQ  1024
#define DIM_NKV 2048
#define C_EPS   0.1f
#define C_SCALE 0.125f   // DH^-0.5

// Async global->LDS path (gfx1250): guarded so either toolchain compiles.
#ifndef TRY_ASYNC_LDS
#define TRY_ASYNC_LDS 1
#endif
#if TRY_ASYNC_LDS && defined(__has_builtin)
#if __has_builtin(__builtin_amdgcn_global_load_async_to_lds_b128) && \
    __has_builtin(__builtin_amdgcn_s_wait_asynccnt)
#define USE_ASYNC_LDS 1
#endif
#endif

#define ASYNC_GPTR(p) ((__attribute__((address_space(1))) gvec4*)(p))
#define ASYNC_LPTR(p) ((__attribute__((address_space(3))) gvec4*)(p))

__device__ __forceinline__ bf16_t f2bf(float f) {
  u32 x = __float_as_uint(f);
  u32 r = x + 0x7FFFu + ((x >> 16) & 1u);   // round-to-nearest-even
  return (bf16_t)(r >> 16);
}
__device__ __forceinline__ float bf2f(bf16_t h) {
  return __uint_as_float(((u32)h) << 16);
}
__device__ __forceinline__ v8f v8f_zero() {
  v8f z = {0.f, 0.f, 0.f, 0.f, 0.f, 0.f, 0.f, 0.f};
  return z;
}

// A-fragment (16x32 bf16, MxK): lane m=l&15, half h=l>>4.
// VGPR p: K = (p>>2)*16 + h*8 + (p&3)*2, two contiguous K per 32-bit read.
__device__ __forceinline__ v16bf frag_a(const bf16_t* base, int ld) {
  int lane = threadIdx.x & 31;
  int m = lane & 15, h = lane >> 4;
  union { u32 u[8]; v16bf v; } f;
#pragma unroll
  for (int p = 0; p < 8; ++p) {
    int k = ((p >> 2) << 4) + (h << 3) + ((p & 3) << 1);
    f.u[p] = *(const u32*)(base + m * ld + k);
  }
  return f.v;
}
// B-fragment (32x16 bf16, KxN) from K-contiguous [n][k] LDS layout:
// lane n=l&15, half h=l>>4; VGPR p: K = (p>>2)*16 + h*8 + (p&3)*2.
__device__ __forceinline__ v16bf frag_b(const bf16_t* base, int ld) {
  int lane = threadIdx.x & 31;
  int n = lane & 15, h = lane >> 4;
  union { u32 u[8]; v16bf v; } f;
#pragma unroll
  for (int p = 0; p < 8; ++p) {
    int k = ((p >> 2) << 4) + (h << 3) + ((p & 3) << 1);
    f.u[p] = *(const u32*)(base + n * ld + k);
  }
  return f.v;
}
__device__ __forceinline__ v8f wmma32(v16bf a, v16bf b, v8f c) {
  return __builtin_amdgcn_wmma_f32_16x16x32_bf16(false, a, false, b, (short)0, c,
                                                 false, false);
}

// ---------------------------------------------------------------------------
// Weight convert + transpose: W (K x N) f32 -> Wt (N x K) bf16.
// ---------------------------------------------------------------------------
__global__ __launch_bounds__(256) void transpose_convert(
    const float* __restrict__ W, bf16_t* __restrict__ Wt, int K, int N) {
  __shared__ float tile[32][33];
  int k0 = blockIdx.x << 5, n0 = blockIdx.y << 5;
  int tx = threadIdx.x & 31, ty = threadIdx.x >> 5;
  for (int i = ty; i < 32; i += 8) tile[i][tx] = W[(size_t)(k0 + i) * N + n0 + tx];
  __syncthreads();
  for (int i = ty; i < 32; i += 8)
    Wt[(size_t)(n0 + i) * K + k0 + tx] = f2bf(tile[tx][i]);
}

// ---------------------------------------------------------------------------
// Router logits: one wave per token, both routers in one pass over x.
// ---------------------------------------------------------------------------
__global__ __launch_bounds__(256) void router_logits(
    const float* __restrict__ x, const float* __restrict__ wq,
    const float* __restrict__ wkv, float* __restrict__ lq,
    float* __restrict__ lkv) {
  int token = blockIdx.x * 8 + (threadIdx.x >> 5);
  int lane = threadIdx.x & 31;
  const float4* xr = (const float4*)(x + (size_t)token * DIM_D);
  const float4* a4 = (const float4*)wq;
  const float4* b4 = (const float4*)wkv;
  float aq = 0.f, akv = 0.f;
#pragma unroll
  for (int j = 0; j < 8; ++j) {
    float4 v = xr[lane + 32 * j];
    float4 a = a4[lane + 32 * j];
    float4 b = b4[lane + 32 * j];
    aq  += v.x * a.x + v.y * a.y + v.z * a.z + v.w * a.w;
    akv += v.x * b.x + v.y * b.y + v.z * b.z + v.w * b.w;
  }
#pragma unroll
  for (int m = 16; m >= 1; m >>= 1) {
    aq  += __shfl_xor(aq,  m, 32);
    akv += __shfl_xor(akv, m, 32);
  }
  if (lane == 0) { lq[token] = aq; lkv[token] = akv; }
}

// ---------------------------------------------------------------------------
// coor_descent + exact top-k (score desc, index asc). One block per (b) row.
// Only scalar `a` is carried: b_i = -relu(s_i + a); score = exp(min(s+a,0)/eps).
// ---------------------------------------------------------------------------
__global__ __launch_bounds__(1024) void coor_select(
    const float* __restrict__ logits, int* __restrict__ sel, int k, int keff) {
  __shared__ float redF[1024];
  __shared__ u32 redU[1024];
  __shared__ u32 scn[1024];
  __shared__ u32 ctr;
  const int t = threadIdx.x;
  const float* row = logits + (size_t)blockIdx.x * DIM_N;
  float sv[8];
#pragma unroll
  for (int i = 0; i < 8; ++i) sv[i] = row[t * 8 + i];

  const float cst = C_EPS * logf((float)keff);
  float a = cst - C_EPS * logf((float)DIM_N);   // first iter: b=-s -> lse(0)=ln N
  for (int it = 0; it < 19; ++it) {
    float tv[8], tmax = -3.4e38f;
#pragma unroll
    for (int i = 0; i < 8; ++i) {
      float bb = fmaxf(sv[i] + a, 0.f);
      tv[i] = (sv[i] - bb) * (1.f / C_EPS);
      tmax = fmaxf(tmax, tv[i]);
    }
    redF[t] = tmax; __syncthreads();
    for (int off = 512; off > 0; off >>= 1) {
      if (t < off) redF[t] = fmaxf(redF[t], redF[t + off]);
      __syncthreads();
    }
    float m = redF[0]; __syncthreads();
    float ssum = 0.f;
#pragma unroll
    for (int i = 0; i < 8; ++i) ssum += __expf(tv[i] - m);
    redF[t] = ssum; __syncthreads();
    for (int off = 512; off > 0; off >>= 1) {
      if (t < off) redF[t] += redF[t + off];
      __syncthreads();
    }
    float lse = m + logf(redF[0]); __syncthreads();
    a = cst - C_EPS * lse;
  }
  u32 u[8];
#pragma unroll
  for (int i = 0; i < 8; ++i)
    u[i] = __float_as_uint(__expf(fminf(sv[i] + a, 0.f) * (1.f / C_EPS)));

  // k-th largest score via bitwise threshold search (scores >= 0 -> bit order).
  u32 T = 0u;
  for (int bit = 31; bit >= 0; --bit) {
    u32 cand = T | (1u << bit);
    u32 cnt = 0;
#pragma unroll
    for (int i = 0; i < 8; ++i) cnt += (u[i] >= cand) ? 1u : 0u;
    redU[t] = cnt; __syncthreads();
    for (int off = 512; off > 0; off >>= 1) {
      if (t < off) redU[t] += redU[t + off];
      __syncthreads();
    }
    if (redU[0] >= (u32)k) T = cand;
    __syncthreads();
  }
  u32 cg = 0, ce = 0;
#pragma unroll
  for (int i = 0; i < 8; ++i) {
    cg += (u[i] > T) ? 1u : 0u;
    ce += (u[i] == T) ? 1u : 0u;
  }
  redU[t] = cg; __syncthreads();
  for (int off = 512; off > 0; off >>= 1) {
    if (t < off) redU[t] += redU[t + off];
    __syncthreads();
  }
  u32 cntG = redU[0]; __syncthreads();
  u32 need = (u32)k - cntG;

  // exclusive scan of per-thread equal counts (index-ascending tie-break)
  scn[t] = ce; __syncthreads();
  for (int off = 1; off < 1024; off <<= 1) {
    u32 v = (t >= off) ? scn[t - off] : 0u;
    __syncthreads();
    scn[t] += v;
    __syncthreads();
  }
  u32 er = scn[t] - ce;
  if (t == 0) ctr = 0u;
  __syncthreads();
  int* selrow = sel + (size_t)blockIdx.x * k;
#pragma unroll
  for (int i = 0; i < 8; ++i) {
    int idx = t * 8 + i;
    if (u[i] > T) {
      u32 p = atomicAdd(&ctr, 1u);
      selrow[p] = idx;
    } else if (u[i] == T) {
      if (er < need) selrow[cntG + er] = idx;
      er++;
    }
  }
}

// ---------------------------------------------------------------------------
// Gather routed token rows of x, convert fp32 -> bf16.
// ---------------------------------------------------------------------------
__global__ __launch_bounds__(256) void gather_bf16(
    const float* __restrict__ x, const int* __restrict__ sel,
    bf16_t* __restrict__ out, int ktok) {
  int b = blockIdx.x / ktok, r = blockIdx.x - b * ktok;
  int idx = sel[(size_t)b * ktok + r];
  const float4* src = (const float4*)(x + ((size_t)b * DIM_N + idx) * DIM_D);
  bf16_t* dst = out + ((size_t)b * ktok + r) * DIM_D;
  int t = threadIdx.x;
  float4 v = src[t];
  *(u32*)&dst[t * 4]     = ((u32)f2bf(v.y) << 16) | f2bf(v.x);
  *(u32*)&dst[t * 4 + 2] = ((u32)f2bf(v.w) << 16) | f2bf(v.z);
}

// ---------------------------------------------------------------------------
// bf16 WMMA GEMM: C[M,N](f32) = A[M,K](bf16,row) * Bt[N,K](bf16,row)^T
// Compile-time N,K -> immediate-offset addressing. 128x128 block tile,
// 8 waves of 32x64, K-step 32, double-buffered LDS staging.
// Staging uses GLOBAL_LOAD_ASYNC_TO_LDS_B128 (ASYNCcnt) when the builtin
// exists; otherwise a register-staged double buffer.
// ---------------------------------------------------------------------------
template <int N, int K>
__global__ __launch_bounds__(256) void gemm_bf16_f32(
    const bf16_t* __restrict__ A, const bf16_t* __restrict__ Bt,
    float* __restrict__ C) {
  __shared__ bf16_t As[2][128][40];
  __shared__ bf16_t Bs[2][128][40];
  const int bm = blockIdx.x * 128, bn = blockIdx.y * 128;
  const int tid = threadIdx.x, wave = tid >> 5, lane = tid & 31;
  const int wm = (wave >> 1) << 5, wn = (wave & 1) << 6;
  const int r = tid >> 1, c = (tid & 1) << 4;
  const bf16_t* gA = A + (size_t)(bm + r) * K + c;   // +k0 per step
  const bf16_t* gB = Bt + (size_t)(bn + r) * K + c;

  v8f acc[2][4];
#pragma unroll
  for (int i = 0; i < 2; ++i)
#pragma unroll
    for (int j = 0; j < 4; ++j) acc[i][j] = v8f_zero();

  constexpr int NSTEP = K / 32;

#if defined(USE_ASYNC_LDS) && defined(__HIP_DEVICE_COMPILE__)
  auto issue = [&](int nb, int k0) {
    __builtin_amdgcn_global_load_async_to_lds_b128(
        ASYNC_GPTR(gA + k0), ASYNC_LPTR(&As[nb][r][c]), 0, 0);
    __builtin_amdgcn_global_load_async_to_lds_b128(
        ASYNC_GPTR(gA + k0 + 8), ASYNC_LPTR(&As[nb][r][c + 8]), 0, 0);
    __builtin_amdgcn_global_load_async_to_lds_b128(
        ASYNC_GPTR(gB + k0), ASYNC_LPTR(&Bs[nb][r][c]), 0, 0);
    __builtin_amdgcn_global_load_async_to_lds_b128(
        ASYNC_GPTR(gB + k0 + 8), ASYNC_LPTR(&Bs[nb][r][c + 8]), 0, 0);
  };
  issue(0, 0);
  __builtin_amdgcn_s_wait_asynccnt(0);
  __syncthreads();

  for (int ks = 0; ks < NSTEP; ++ks) {
    const int cur = ks & 1;
    if (ks + 1 < NSTEP) issue(cur ^ 1, (ks + 1) * 32);
    v16bf af[2], bfv[4];
#pragma unroll
    for (int i = 0; i < 2; ++i) af[i] = frag_a(&As[cur][wm + (i << 4)][0], 40);
#pragma unroll
    for (int j = 0; j < 4; ++j) bfv[j] = frag_b(&Bs[cur][wn + (j << 4)][0], 40);
#pragma unroll
    for (int i = 0; i < 2; ++i)
#pragma unroll
      for (int j = 0; j < 4; ++j) acc[i][j] = wmma32(af[i], bfv[j], acc[i][j]);
    __builtin_amdgcn_s_wait_asynccnt(0);
    __syncthreads();
  }
#else
  // Fallback: register-staged double buffer (prefetch overlaps WMMA).
  {
    uint4 a0 = ((const uint4*)gA)[0], a1 = ((const uint4*)gA)[1];
    uint4 b0 = ((const uint4*)gB)[0], b1 = ((const uint4*)gB)[1];
    *(uint4*)&As[0][r][c] = a0; *(uint4*)&As[0][r][c + 8] = a1;
    *(uint4*)&Bs[0][r][c] = b0; *(uint4*)&Bs[0][r][c + 8] = b1;
  }
  __syncthreads();
  for (int ks = 0; ks < NSTEP; ++ks) {
    const int cur = ks & 1;
    uint4 a0, a1, b0, b1;
    const bool pre = (ks + 1 < NSTEP);
    if (pre) {
      const bf16_t* pa = gA + (ks + 1) * 32;
      const bf16_t* pb = gB + (ks + 1) * 32;
      a0 = ((const uint4*)pa)[0]; a1 = ((const uint4*)pa)[1];
      b0 = ((const uint4*)pb)[0]; b1 = ((const uint4*)pb)[1];
    }
    v16bf af[2], bfv[4];
#pragma unroll
    for (int i = 0; i < 2; ++i) af[i] = frag_a(&As[cur][wm + (i << 4)][0], 40);
#pragma unroll
    for (int j = 0; j < 4; ++j) bfv[j] = frag_b(&Bs[cur][wn + (j << 4)][0], 40);
#pragma unroll
    for (int i = 0; i < 2; ++i)
#pragma unroll
      for (int j = 0; j < 4; ++j) acc[i][j] = wmma32(af[i], bfv[j], acc[i][j]);
    if (pre) {
      *(uint4*)&As[cur ^ 1][r][c] = a0; *(uint4*)&As[cur ^ 1][r][c + 8] = a1;
      *(uint4*)&Bs[cur ^ 1][r][c] = b0; *(uint4*)&Bs[cur ^ 1][r][c + 8] = b1;
    }
    __syncthreads();
  }
#endif

  const int half = lane >> 4, col = lane & 15;
#pragma unroll
  for (int i = 0; i < 2; ++i) {
    float* crow = C + (size_t)(bm + wm + (i << 4) + (half << 3)) * N +
                  (bn + wn + col);
#pragma unroll
    for (int j = 0; j < 4; ++j)
#pragma unroll
      for (int r8 = 0; r8 < 8; ++r8)
        crow[r8 * N + (j << 4)] = acc[i][j][r8];   // const strides -> imm offs
  }
}

// ---------------------------------------------------------------------------
// Rotary on q projection (router score forward == 1.0, so no rescale).
// ---------------------------------------------------------------------------
__global__ __launch_bounds__(256) void rotary_gather_q(
    const float* __restrict__ proj, const float* __restrict__ rope,
    const int* __restrict__ sel, bf16_t* __restrict__ out) {
  __shared__ float qs[1024];
  __shared__ float cs[64], sn[64];
  int row = blockIdx.x;                 // b*NQ + r
  int b = row >> 10, r = row & 1023;
  int idx = sel[(size_t)b * DIM_NQ + r];
  int t = threadIdx.x;
  float4 v = ((const float4*)(proj + (size_t)row * DIM_D))[t];
  qs[t * 4] = v.x; qs[t * 4 + 1] = v.y; qs[t * 4 + 2] = v.z; qs[t * 4 + 3] = v.w;
  if (t < 64) {
    float p = rope[(size_t)idx * DIM_DH + t];
    cs[t] = cosf(p); sn[t] = sinf(p);
  }
  __syncthreads();
  bf16_t ov[4];
#pragma unroll
  for (int i = 0; i < 4; ++i) {
    int d = t * 4 + i, dh = d & 63;
    float rot = (dh < 32) ? -qs[d + 32] : qs[d - 32];
    ov[i] = f2bf(qs[d] * cs[dh] + rot * sn[dh]);
  }
  bf16_t* dst = out + (size_t)row * DIM_D;
  *(u32*)&dst[t * 4]     = ((u32)ov[1] << 16) | ov[0];
  *(u32*)&dst[t * 4 + 2] = ((u32)ov[3] << 16) | ov[2];
}

// kv projection: row layout (2, H, DH) -> k=[0,1024) gets rotary, v=[1024,2048).
__global__ __launch_bounds__(256) void rotary_gather_kv(
    const float* __restrict__ proj, const float* __restrict__ rope,
    const int* __restrict__ sel, bf16_t* __restrict__ kout,
    bf16_t* __restrict__ vout) {
  __shared__ float ks[1024];
  __shared__ float cs[64], sn[64];
  int row = blockIdx.x;                 // b*NKV + r
  int b = row >> 11, r = row & 2047;
  int idx = sel[(size_t)b * DIM_NKV + r];
  int t = threadIdx.x;
  const float* base = proj + (size_t)row * 2048;
  float4 kv = ((const float4*)base)[t];
  ks[t * 4] = kv.x; ks[t * 4 + 1] = kv.y; ks[t * 4 + 2] = kv.z; ks[t * 4 + 3] = kv.w;
  float4 vv = ((const float4*)(base + 1024))[t];
  if (t < 64) {
    float p = rope[(size_t)idx * DIM_DH + t];
    cs[t] = cosf(p); sn[t] = sinf(p);
  }
  __syncthreads();
  bf16_t kk[4];
#pragma unroll
  for (int i = 0; i < 4; ++i) {
    int d = t * 4 + i, dh = d & 63;
    float rot = (dh < 32) ? -ks[d + 32] : ks[d - 32];
    kk[i] = f2bf(ks[d] * cs[dh] + rot * sn[dh]);
  }
  bf16_t* kd = kout + (size_t)row * DIM_D;
  *(u32*)&kd[t * 4]     = ((u32)kk[1] << 16) | kk[0];
  *(u32*)&kd[t * 4 + 2] = ((u32)kk[3] << 16) | kk[2];
  bf16_t* vd = vout + (size_t)row * DIM_D;
  *(u32*)&vd[t * 4]     = ((u32)f2bf(vv.y) << 16) | f2bf(vv.x);
  *(u32*)&vd[t * 4 + 2] = ((u32)f2bf(vv.w) << 16) | f2bf(vv.z);
}

// ---------------------------------------------------------------------------
// Flash attention per (b, h, 64-q tile). 4 waves x 16 q-rows, 64-wide kv tiles.
// Null KV folded into softmax init: m0 = q.nk*scale, l0 = 1, O0 = nv.
// ---------------------------------------------------------------------------
__global__ __launch_bounds__(128) void attention_kernel(
    const bf16_t* __restrict__ qb, const bf16_t* __restrict__ kb,
    const bf16_t* __restrict__ vb, const float* __restrict__ null_kv,
    bf16_t* __restrict__ ob) {
  __shared__ bf16_t Qs[64][72];
  __shared__ bf16_t Ks[64][72];
  __shared__ bf16_t Vt[64][72];     // transposed: [dh][kv]
  __shared__ bf16_t Ps[4][16][72];  // per-wave P tile
  __shared__ float nk[64], nv[64];

  const int qt = blockIdx.x & 15;
  const int h  = (blockIdx.x >> 4) & 15;
  const int b  = blockIdx.x >> 8;
  const int tid = threadIdx.x, w = tid >> 5, lane = tid & 31;
  const int half = lane >> 4, col = lane & 15;

  {
    int i = tid >> 1, c = (tid & 1) << 5;
    const uint4* src = (const uint4*)(
        qb + ((size_t)(b * DIM_NQ + qt * 64 + i) * DIM_H + h) * DIM_DH + c);
    uint4 x0 = src[0], x1 = src[1], x2 = src[2], x3 = src[3];
    *(uint4*)&Qs[i][c]      = x0; *(uint4*)&Qs[i][c + 8]  = x1;
    *(uint4*)&Qs[i][c + 16] = x2; *(uint4*)&Qs[i][c + 24] = x3;
  }
  if (tid < 64) nk[tid] = null_kv[(size_t)h * DIM_DH + tid];
  else          nv[tid - 64] = null_kv[(size_t)(DIM_H + h) * DIM_DH + (tid - 64)];
  __syncthreads();

  float ms[8], ls[8];
  v8f acc[4];
#pragma unroll
  for (int rr = 0; rr < 8; ++rr) {
    int qrow = (w << 4) + (half << 3) + rr;
    float s = 0.f;
    for (int d = 0; d < 64; ++d) s += bf2f(Qs[qrow][d]) * nk[d];
    ms[rr] = s * C_SCALE;
    ls[rr] = 1.f;
  }
#pragma unroll
  for (int j = 0; j < 4; ++j) {
    float nvv = nv[(j << 4) + col];
#pragma unroll
    for (int rr = 0; rr < 8; ++rr) acc[j][rr] = nvv;
  }

  for (int kt = 0; kt < DIM_NKV / 64; ++kt) {
    {
      int i = tid >> 1, c = (tid & 1) << 5;
      const uint4* ksrc = (const uint4*)(
          kb + ((size_t)(b * DIM_NKV + kt * 64 + i) * DIM_H + h) * DIM_DH + c);
      uint4 x0 = ksrc[0], x1 = ksrc[1], x2 = ksrc[2], x3 = ksrc[3];
      *(uint4*)&Ks[i][c]      = x0; *(uint4*)&Ks[i][c + 8]  = x1;
      *(uint4*)&Ks[i][c + 16] = x2; *(uint4*)&Ks[i][c + 24] = x3;
      const uint4* vsrc = (const uint4*)(
          vb + ((size_t)(b * DIM_NKV + kt * 64 + i) * DIM_H + h) * DIM_DH + c);
      union { uint4 q[4]; bf16_t e[32]; } vy;
      vy.q[0] = vsrc[0]; vy.q[1] = vsrc[1]; vy.q[2] = vsrc[2]; vy.q[3] = vsrc[3];
#pragma unroll
      for (int d = 0; d < 32; ++d) Vt[c + d][i] = vy.e[d];
    }
    __syncthreads();

    // S = Q K^T
    v16bf aq0 = frag_a(&Qs[w << 4][0], 72);
    v16bf aq1 = frag_a(&Qs[w << 4][32], 72);
    v8f sfrag[4];
#pragma unroll
    for (int j = 0; j < 4; ++j) {
      v16bf bk0 = frag_b(&Ks[j << 4][0], 72);
      v16bf bk1 = frag_b(&Ks[j << 4][32], 72);
      v8f z = v8f_zero();
      z = wmma32(aq0, bk0, z);
      z = wmma32(aq1, bk1, z);
      sfrag[j] = z;
    }
    // online softmax (rows live in 16-lane halves; xor-shuffle 1/2/4/8)
    float pv[4][8];
#pragma unroll
    for (int rr = 0; rr < 8; ++rr) {
      float mx = -3.4e38f;
#pragma unroll
      for (int j = 0; j < 4; ++j) {
        sfrag[j][rr] *= C_SCALE;
        mx = fmaxf(mx, sfrag[j][rr]);
      }
#pragma unroll
      for (int d = 1; d < 16; d <<= 1) mx = fmaxf(mx, __shfl_xor(mx, d, 32));
      float mnew = fmaxf(ms[rr], mx);
      float corr = __expf(ms[rr] - mnew);
      float rsum = 0.f;
#pragma unroll
      for (int j = 0; j < 4; ++j) {
        float p = __expf(sfrag[j][rr] - mnew);
        pv[j][rr] = p; rsum += p;
      }
#pragma unroll
      for (int d = 1; d < 16; d <<= 1) rsum += __shfl_xor(rsum, d, 32);
      ls[rr] = ls[rr] * corr + rsum;
      ms[rr] = mnew;
#pragma unroll
      for (int j = 0; j < 4; ++j) acc[j][rr] *= corr;
    }
#pragma unroll
    for (int j = 0; j < 4; ++j)
#pragma unroll
      for (int rr = 0; rr < 8; ++rr)
        Ps[w][(half << 3) + rr][(j << 4) + col] = f2bf(pv[j][rr]);
    __syncthreads();

    // O += P V
    v16bf ap0 = frag_a(&Ps[w][0][0], 72);
    v16bf ap1 = frag_a(&Ps[w][0][32], 72);
#pragma unroll
    for (int j = 0; j < 4; ++j) {
      v16bf bv0 = frag_b(&Vt[j << 4][0], 72);
      v16bf bv1 = frag_b(&Vt[j << 4][32], 72);
      acc[j] = wmma32(ap0, bv0, acc[j]);
      acc[j] = wmma32(ap1, bv1, acc[j]);
    }
    __syncthreads();
  }
#pragma unroll
  for (int j = 0; j < 4; ++j)
#pragma unroll
    for (int rr = 0; rr < 8; ++rr) {
      int qrow = qt * 64 + (w << 4) + (half << 3) + rr;
      float o = acc[j][rr] / ls[rr];
      ob[((size_t)(b * DIM_NQ + qrow) * DIM_H + h) * DIM_DH + (j << 4) + col] =
          f2bf(o);
    }
}

// ---------------------------------------------------------------------------
// Base fill + scatter.
// ---------------------------------------------------------------------------
__global__ __launch_bounds__(256) void fill_base(
    const float* __restrict__ nt, float* __restrict__ out) {
  float4 v = ((const float4*)nt)[threadIdx.x];
  ((float4*)out)[(size_t)blockIdx.x * 256 + threadIdx.x] = v;
}

__global__ __launch_bounds__(256) void scatter_rows(
    const float* __restrict__ proj, const int* __restrict__ sel,
    float* __restrict__ out) {
  int row = blockIdx.x;
  int b = row >> 10, r = row & 1023;
  int idx = sel[(size_t)b * DIM_NQ + r];
  float4 v = ((const float4*)(proj + (size_t)row * DIM_D))[threadIdx.x];
  ((float4*)(out + ((size_t)b * DIM_N + idx) * DIM_D))[threadIdx.x] = v;
}

// ---------------------------------------------------------------------------
extern "C" void kernel_launch(void* const* d_in, const int* in_sizes, int n_in,
                              void* d_out, int out_size, void* d_ws,
                              size_t ws_size, hipStream_t stream) {
  const float* x       = (const float*)d_in[0];
  const float* rope    = (const float*)d_in[1];
  const float* wrq     = (const float*)d_in[2];
  const float* wrkv    = (const float*)d_in[3];
  const float* w_q     = (const float*)d_in[4];
  const float* w_kv    = (const float*)d_in[5];
  const float* w_out   = (const float*)d_in[6];
  const float* nullkv  = (const float*)d_in[7];
  const float* nulltok = (const float*)d_in[8];
  float* out = (float*)d_out;

  char* ws = (char*)d_ws;
  size_t off = 0;
  auto alloc = [&](size_t bytes) -> void* {
    off = (off + 255) & ~(size_t)255;
    void* p = ws + off;
    off += bytes;
    return p;
  };

  bf16_t* wqT    = (bf16_t*)alloc((size_t)1024 * 1024 * 2);
  bf16_t* wkvT   = (bf16_t*)alloc((size_t)2048 * 1024 * 2);
  bf16_t* woutT  = (bf16_t*)alloc((size_t)1024 * 1024 * 2);
  float*  logQ   = (float*)alloc((size_t)DIM_B * DIM_N * 4);
  float*  logKV  = (float*)alloc((size_t)DIM_B * DIM_N * 4);
  int*    selQ   = (int*)alloc((size_t)DIM_B * DIM_NQ * 4);
  int*    selKV  = (int*)alloc((size_t)DIM_B * DIM_NKV * 4);
  bf16_t* xq     = (bf16_t*)alloc((size_t)DIM_B * DIM_NQ * DIM_D * 2);
  bf16_t* xkv    = (bf16_t*)alloc((size_t)DIM_B * DIM_NKV * DIM_D * 2);
  bf16_t* qbf    = (bf16_t*)alloc((size_t)DIM_B * DIM_NQ * DIM_D * 2);
  bf16_t* kbf    = (bf16_t*)alloc((size_t)DIM_B * DIM_NKV * DIM_D * 2);
  bf16_t* vbf    = (bf16_t*)alloc((size_t)DIM_B * DIM_NKV * DIM_D * 2);
  bf16_t* aout   = (bf16_t*)alloc((size_t)DIM_B * DIM_NQ * DIM_D * 2);
  float*  kvproj = (float*)alloc((size_t)DIM_B * DIM_NKV * 2048 * 4);
  float*  qproj  = (float*)alloc((size_t)DIM_B * DIM_NQ * DIM_D * 4);
  float*  oproj  = kvproj;  // kvproj is dead after rotary_gather_kv

  dim3 b256(256), b1024(1024), b128(128);

  transpose_convert<<<dim3(32, 32), b256, 0, stream>>>(w_q, wqT, 1024, 1024);
  transpose_convert<<<dim3(32, 64), b256, 0, stream>>>(w_kv, wkvT, 1024, 2048);
  transpose_convert<<<dim3(32, 32), b256, 0, stream>>>(w_out, woutT, 1024, 1024);

  router_logits<<<dim3((DIM_B * DIM_N) / 8), b256, 0, stream>>>(x, wrq, wrkv,
                                                                logQ, logKV);
  coor_select<<<dim3(DIM_B), b1024, 0, stream>>>(logQ, selQ, DIM_NQ, 1152);
  coor_select<<<dim3(DIM_B), b1024, 0, stream>>>(logKV, selKV, DIM_NKV, 2304);

  gather_bf16<<<dim3(DIM_B * DIM_NQ), b256, 0, stream>>>(x, selQ, xq, DIM_NQ);
  gather_bf16<<<dim3(DIM_B * DIM_NKV), b256, 0, stream>>>(x, selKV, xkv, DIM_NKV);

  gemm_bf16_f32<1024, 1024><<<dim3(32, 8), b256, 0, stream>>>(xq, wqT, qproj);
  gemm_bf16_f32<2048, 1024><<<dim3(64, 16), b256, 0, stream>>>(xkv, wkvT,
                                                               kvproj);

  rotary_gather_q<<<dim3(DIM_B * DIM_NQ), b256, 0, stream>>>(qproj, rope, selQ,
                                                             qbf);
  rotary_gather_kv<<<dim3(DIM_B * DIM_NKV), b256, 0, stream>>>(kvproj, rope,
                                                               selKV, kbf, vbf);

  attention_kernel<<<dim3(DIM_B * DIM_H * (DIM_NQ / 64)), b128, 0, stream>>>(
      qbf, kbf, vbf, nullkv, aout);

  gemm_bf16_f32<1024, 1024><<<dim3(32, 8), b256, 0, stream>>>(aout, woutT,
                                                              oproj);

  fill_base<<<dim3(DIM_B * DIM_N), b256, 0, stream>>>(nulltok, out);
  scatter_rows<<<dim3(DIM_B * DIM_NQ), b256, 0, stream>>>(oproj, selQ, out);
}
